// SimAttention_23347442221590
// MI455X (gfx1250) — compile-verified, matrix operand order checked by reading, not run on hardware
//
#include <hip/hip_runtime.h>
#include <hip/hip_bf16.h>

#define BB 64
#define LL 512
#define DD 1024
#define HH 16
#define DHE 64

typedef __attribute__((ext_vector_type(16))) __bf16          v16bf;
typedef __attribute__((ext_vector_type(8)))  float           v8f;
typedef __attribute__((ext_vector_type(16))) unsigned short  v16u;
typedef __attribute__((ext_vector_type(8)))  unsigned short  v8u;
typedef __attribute__((ext_vector_type(4)))  unsigned int    v4ui;

static __device__ __forceinline__ unsigned short f2bf(float f) {
  unsigned int u = __builtin_bit_cast(unsigned int, f);
  u += 0x7FFFu + ((u >> 16) & 1u);            // round-to-nearest-even
  return (unsigned short)(u >> 16);
}

static __device__ __forceinline__ v8f wmma_bf16(v16u a, v16u b, v8f c) {
  return __builtin_amdgcn_wmma_f32_16x16x32_bf16(
      false, __builtin_bit_cast(v16bf, a),
      false, __builtin_bit_cast(v16bf, b),
      (short)0, c, false, false);
}

static __device__ __forceinline__ void ld8(v16u& d, int base, const unsigned short* p) {
  v8u t = *(const v8u*)p;                     // ds_load_b128 / global_load_b128
#pragma unroll
  for (int i = 0; i < 8; ++i) d[base + i] = t[i];
}

static __device__ __forceinline__ v8u pack8(float4 a, float4 b) {
  v4ui w;
  w[0] = (unsigned)f2bf(a.x) | ((unsigned)f2bf(a.y) << 16);
  w[1] = (unsigned)f2bf(a.z) | ((unsigned)f2bf(a.w) << 16);
  w[2] = (unsigned)f2bf(b.x) | ((unsigned)f2bf(b.y) << 16);
  w[3] = (unsigned)f2bf(b.z) | ((unsigned)f2bf(b.w) << 16);
  return __builtin_bit_cast(v8u, w);
}

// ---------------------------------------------------------------------------
// Kernel 1: fused flash attention per (b, h, 128-query tile)
//   - double-buffered K/V LDS tiles, register-staged global fetch (ping-pong)
//   - row-sum of P computed with an extra WMMA against a ones matrix
// ---------------------------------------------------------------------------
#define KSTR 72   // K_lds row stride (bf16 elems): 144B, 16B-aligned, bank-spread
#define VTSTR 40  // VT_lds row stride: 80B
#define PSTR 40   // P scratch row stride: 80B

__global__ __launch_bounds__(256) void attn_kernel(
    const float* __restrict__ q, const float* __restrict__ k,
    const float* __restrict__ v, const float* __restrict__ sw,
    const int* __restrict__ mask, unsigned short* __restrict__ hidden)
{
  __shared__ unsigned short k_lds[2][32 * KSTR];    // K chunk  [key][dh]
  __shared__ unsigned short vt_lds[2][64 * VTSTR];  // V chunk transposed [dh][key]
  __shared__ unsigned short p_lds[8 * 16 * PSTR];   // per-wave P relayout scratch

  const int bh   = blockIdx.x;
  const int b    = bh / HH;
  const int h    = bh % HH;
  const int tid  = threadIdx.x;
  const int wave = tid >> 5;
  const int lane = tid & 31;
  const int r    = lane & 15;      // A row / B,C column
  const int hh   = lane >> 4;      // K-half / M-half selector
  const int qbase = blockIdx.y * 128 + wave * 16;

  // cooperative-loader coordinates
  const int ckey = tid >> 3;           // 0..31
  const int cd0  = (tid & 7) * 8;      // 0..56

  // ---- preload Q (16x64) as two bf16 A-operands (K = 0..31 and 32..63) ----
  const float* qrow = q + ((size_t)(b * LL + qbase + r) * DD) + h * DHE;
  v16u aq0, aq1;
  {
    const float4* p = (const float4*)(qrow + hh * 8);
    v8u t = pack8(p[0], p[1]);
#pragma unroll
    for (int i = 0; i < 8; ++i) aq0[i] = t[i];
    p = (const float4*)(qrow + 16 + hh * 8);
    t = pack8(p[0], p[1]);
#pragma unroll
    for (int i = 0; i < 8; ++i) aq0[8 + i] = t[i];
    p = (const float4*)(qrow + 32 + hh * 8);
    t = pack8(p[0], p[1]);
#pragma unroll
    for (int i = 0; i < 8; ++i) aq1[i] = t[i];
    p = (const float4*)(qrow + 48 + hh * 8);
    t = pack8(p[0], p[1]);
#pragma unroll
    for (int i = 0; i < 8; ++i) aq1[8 + i] = t[i];
  }

  // all-ones bf16 B operand for row-sum WMMA
  v16u onesv;
#pragma unroll
  for (int i = 0; i < 16; ++i) onesv[i] = 0x3F80;

  // per-row learned scale (row of C element j is qbase + j + 8*hh)
  float swv[8];
#pragma unroll
  for (int j = 0; j < 8; ++j)
    swv[j] = 0.125f * sw[h * LL + qbase + j + 8 * hh];

  v8f   o[4] = {};                    // O accumulators: 4 dh-tiles of 16x16
  float m[8], l[8];
#pragma unroll
  for (int j = 0; j < 8; ++j) { m[j] = -INFINITY; l[j] = 0.0f; }

  // register-staged fetch of one K/V chunk (8 f32 each per thread)
  float4 kf0, kf1, vf0, vf1;
  auto fetch = [&](int ch) {
    const size_t go = ((size_t)(b * LL + ch + ckey) * DD) + h * DHE + cd0;
    const float4* kp = (const float4*)(k + go);
    kf0 = kp[0]; kf1 = kp[1];
    const float4* vp = (const float4*)(v + go);
    vf0 = vp[0]; vf1 = vp[1];
  };
  auto store_lds = [&](int buf) {
    *(v8u*)&k_lds[buf][ckey * KSTR + cd0] = pack8(kf0, kf1);
    float vv[8] = {vf0.x, vf0.y, vf0.z, vf0.w, vf1.x, vf1.y, vf1.z, vf1.w};
#pragma unroll
    for (int i = 0; i < 8; ++i)          // transpose into [dh][key]
      vt_lds[buf][(cd0 + i) * VTSTR + ckey] = f2bf(vv[i]);
  };

  fetch(0);
  store_lds(0);

  for (int it = 0; it < LL / 32; ++it) {
    const int ch  = it * 32;
    const int cur = it & 1;
    if (it + 1 < LL / 32) fetch(ch + 32);       // issue next chunk's loads early
    __syncthreads();                            // buf[cur] ready for everyone
    if (it + 1 < LL / 32) store_lds(cur ^ 1);   // fill other buffer

    // ---- S = Q * K^T  (two 16-key tiles, K-dim 64 = 2 WMMAs each) ----
    const unsigned short* kbuf = k_lds[cur];
    v8f s0 = {}, s1 = {};
    {
      const unsigned short* kr = &kbuf[r * KSTR];
      v16u b0, b1;
      ld8(b0, 0, kr + hh * 8);       ld8(b0, 8, kr + 16 + hh * 8);
      ld8(b1, 0, kr + 32 + hh * 8);  ld8(b1, 8, kr + 48 + hh * 8);
      s0 = wmma_bf16(aq0, b0, s0);
      s0 = wmma_bf16(aq1, b1, s0);
      kr = &kbuf[(16 + r) * KSTR];
      ld8(b0, 0, kr + hh * 8);       ld8(b0, 8, kr + 16 + hh * 8);
      ld8(b1, 0, kr + 32 + hh * 8);  ld8(b1, 8, kr + 48 + hh * 8);
      s1 = wmma_bf16(aq0, b0, s1);
      s1 = wmma_bf16(aq1, b1, s1);
    }

    // ---- scale, mask, online softmax (max via 16-lane shuffles) ----
    const int mk0 = mask[b * LL + ch + r];
    const int mk1 = mask[b * LL + ch + 16 + r];
    float alpha[8];
#pragma unroll
    for (int j = 0; j < 8; ++j) {
      float x0 = s0[j] * swv[j]; if (mk0 == 0) x0 = -1e9f;
      float x1 = s1[j] * swv[j]; if (mk1 == 0) x1 = -1e9f;
      float cm = fmaxf(x0, x1);
      cm = fmaxf(cm, __shfl_xor(cm, 1));
      cm = fmaxf(cm, __shfl_xor(cm, 2));
      cm = fmaxf(cm, __shfl_xor(cm, 4));
      cm = fmaxf(cm, __shfl_xor(cm, 8));      // row max within 16-lane group
      const float mn = fmaxf(m[j], cm);
      alpha[j] = __expf(m[j] - mn);
      m[j] = mn;
      const float e0 = __expf(x0 - mn);
      const float e1 = __expf(x1 - mn);
      unsigned short* pr = &p_lds[(wave * 16 + j + 8 * hh) * PSTR];
      pr[r]      = f2bf(e0);                  // C-layout -> LDS (row-major)
      pr[16 + r] = f2bf(e1);
    }

    // ---- read P back in A layout ----
    const unsigned short* prow = &p_lds[(wave * 16 + r) * PSTR];
    v16u ap;
    ld8(ap, 0, prow + hh * 8);
    ld8(ap, 8, prow + 16 + hh * 8);

    // ---- row sums of P via WMMA against ones (replaces 32 ds_bpermute) ----
    v8f zc = {};
    v8f rs = wmma_bf16(ap, onesv, zc);
#pragma unroll
    for (int j = 0; j < 8; ++j) l[j] = l[j] * alpha[j] + rs[j];

#pragma unroll
    for (int nt = 0; nt < 4; ++nt)
#pragma unroll
      for (int j = 0; j < 8; ++j) o[nt][j] *= alpha[j];

    // ---- O += P * V ----
    const unsigned short* vbuf = vt_lds[cur];
#pragma unroll
    for (int nt = 0; nt < 4; ++nt) {
      const unsigned short* vr = &vbuf[(nt * 16 + r) * VTSTR];
      v16u bv;
      ld8(bv, 0, vr + hh * 8);
      ld8(bv, 8, vr + 16 + hh * 8);
      o[nt] = wmma_bf16(ap, bv, o[nt]);
    }
  }

  // ---- normalize and write bf16 hidden [B,L,D] ----
  float invl[8];
#pragma unroll
  for (int j = 0; j < 8; ++j) invl[j] = 1.0f / l[j];
#pragma unroll
  for (int nt = 0; nt < 4; ++nt)
#pragma unroll
    for (int j = 0; j < 8; ++j) {
      const int row = qbase + j + 8 * hh;
      hidden[(size_t)(b * LL + row) * DD + h * DHE + nt * 16 + r] =
          f2bf(o[nt][j] * invl[j]);
    }
}

// ---------------------------------------------------------------------------
// Kernel 2: out = hidden(bf16) @ W^T + b  (M=B*L=32768, N=K=1024)
//   - double-buffered W tile, register-staged fetch
// ---------------------------------------------------------------------------
#define WSTR 40

__global__ __launch_bounds__(256) void proj_kernel(
    const unsigned short* __restrict__ hid, const float* __restrict__ W,
    const float* __restrict__ bias, float* __restrict__ out)
{
  __shared__ unsigned short w_lds[2][64 * WSTR];   // W tile [n][k] bf16

  const int nbase = blockIdx.x * 64;
  const int mbase = blockIdx.y * 128;
  const int tid  = threadIdx.x;
  const int wave = tid >> 5;
  const int lane = tid & 31;
  const int r    = lane & 15;
  const int hh   = lane >> 4;

  // cooperative W-loader coordinates: 64 rows x (4 threads x 8 elems)
  const int wn  = tid >> 2;            // 0..63
  const int wkk = (tid & 3) * 8;       // 0..24
  const float* wbase = W + (size_t)(nbase + wn) * DD + wkk;

  v8f acc[4] = {};
  const unsigned short* arow = hid + (size_t)(mbase + wave * 16 + r) * DD;

  float4 wf0, wf1;
  {
    const float4* wp = (const float4*)wbase;
    wf0 = wp[0]; wf1 = wp[1];
  }
  *(v8u*)&w_lds[0][wn * WSTR + wkk] = pack8(wf0, wf1);

  for (int it = 0; it < DD / 32; ++it) {
    const int kc  = it * 32;
    const int cur = it & 1;
    if (it + 1 < DD / 32) {                      // issue next W chunk loads early
      const float4* wp = (const float4*)(wbase + (it + 1) * 32);
      wf0 = wp[0]; wf1 = wp[1];
    }
    __syncthreads();                             // w_lds[cur] ready
    if (it + 1 < DD / 32)
      *(v8u*)&w_lds[cur ^ 1][wn * WSTR + wkk] = pack8(wf0, wf1);

    v16u a;
    ld8(a, 0, arow + kc + hh * 8);
    ld8(a, 8, arow + kc + 16 + hh * 8);
#pragma unroll
    for (int nt = 0; nt < 4; ++nt) {
      const unsigned short* wr = &w_lds[cur][(nt * 16 + r) * WSTR];
      v16u bo;
      ld8(bo, 0, wr + hh * 8);
      ld8(bo, 8, wr + 16 + hh * 8);
      acc[nt] = wmma_bf16(a, bo, acc[nt]);
    }
  }

#pragma unroll
  for (int nt = 0; nt < 4; ++nt) {
    const float bv = bias[nbase + nt * 16 + r];
#pragma unroll
    for (int j = 0; j < 8; ++j)
      out[(size_t)(mbase + wave * 16 + j + 8 * hh) * DD + nbase + nt * 16 + r] =
          acc[nt][j] + bv;
  }
}

// ---------------------------------------------------------------------------
extern "C" void kernel_launch(void* const* d_in, const int* in_sizes, int n_in,
                              void* d_out, int out_size, void* d_ws, size_t ws_size,
                              hipStream_t stream) {
  (void)in_sizes; (void)n_in; (void)out_size; (void)ws_size;
  const float* q    = (const float*)d_in[0];
  const float* k    = (const float*)d_in[1];
  const float* v    = (const float*)d_in[2];
  const float* sw   = (const float*)d_in[3];
  const float* W    = (const float*)d_in[4];
  const float* bias = (const float*)d_in[5];
  const int*   mask = (const int*)d_in[6];

  unsigned short* hidden = (unsigned short*)d_ws;   // 64*512*1024 bf16 = 64 MiB

  attn_kernel<<<dim3(BB * HH, LL / 128), 256, 0, stream>>>(q, k, v, sw, mask, hidden);
  proj_kernel<<<dim3(DD / 64, (BB * LL) / 128), 256, 0, stream>>>(hidden, W, bias,
                                                                  (float*)d_out);
}
